// GNN_67061619359766
// MI455X (gfx1250) — matrix operand order repaired and checked by reading, not compiled, not run on hardware
//
#include <hip/hip_runtime.h>
#include <hip/hip_bf16.h>
#include <math.h>

// HGT forward for MI455X (gfx1250, wave32, WMMA).
// Dense linears: f16 WMMA 16x16x32, f32 accumulate.
// Weights pre-packed into B-fragment layout -> 2x b128 loads per fragment.

typedef __attribute__((ext_vector_type(16))) _Float16 v16h;
typedef __attribute__((ext_vector_type(8)))  _Float16 v8h;
typedef __attribute__((ext_vector_type(8)))  float    v8f;

#define HGT_N   20000
#define HGT_E   150000
#define HGT_HID 256
#define HGT_T   4
#define HGT_R   8
#define HGT_H   8
#define HGT_DK  32
#define HGT_ML  240

// ---------------- elementwise helpers ----------------

__global__ void k_cvt_f16(const float* __restrict__ s, _Float16* __restrict__ d, int n) {
  int i = blockIdx.x * blockDim.x + threadIdx.x;
  int st = gridDim.x * blockDim.x;
  for (; i < n; i += st) d[i] = (_Float16)s[i];
}

__global__ void k_gelu_cvt(const float* __restrict__ s, _Float16* __restrict__ d, int n) {
  int i = blockIdx.x * blockDim.x + threadIdx.x;
  int st = gridDim.x * blockDim.x;
  for (; i < n; i += st) {
    float x = s[i];
    float g = 0.5f * x * (1.0f + erff(x * 0.70710678118654752f));
    d[i] = (_Float16)g;
  }
}

__global__ void k_zero32(unsigned* __restrict__ p, int n) {
  int i = blockIdx.x * blockDim.x + threadIdx.x;
  int st = gridDim.x * blockDim.x;
  for (; i < n; i += st) p[i] = 0u;
}

// ---------------- weight pack: f32 [256][256] -> f16 B-fragment layout ---------
// Packed layout per 256x256 matrix (65536 halves):
//   slot = (tn*8 + kb)*32 + lane ; halves i=0..15 contiguous.
//   source element: k = kb*32 + (lane>>4)*16 + i ; col = tn*16 + (lane&15)
// A lane's whole B fragment = 32 contiguous bytes (32B aligned).

__global__ void k_pack_b(const float* __restrict__ s, _Float16* __restrict__ d, int nmats) {
  int total = nmats * 65536;
  int idx = blockIdx.x * blockDim.x + threadIdx.x;
  int st  = gridDim.x * blockDim.x;
  for (; idx < total; idx += st) {
    int c    = idx >> 16;
    int rem  = idx & 65535;
    int i    = rem & 15;
    int slot = rem >> 4;
    int lane = slot & 31;
    int kbtn = slot >> 5;
    int kb   = kbtn & 7;
    int tn   = kbtn >> 3;
    int k    = kb * 32 + ((lane >> 4) << 4) + i;
    int col  = tn * 16 + (lane & 15);
    d[idx] = (_Float16)s[((size_t)c << 16) + (size_t)k * 256 + col];
  }
}

__device__ __forceinline__ v16h load_a_frag(const _Float16* ap) {
  v8h lo = *(const v8h*)(ap);
  v8h hi = *(const v8h*)(ap + 8);
  return __builtin_shufflevector(lo, hi, 0, 1, 2, 3, 4, 5, 6, 7,
                                 8, 9, 10, 11, 12, 13, 14, 15);
}

// ---------------- WMMA GEMM (K=256, Nout=256), untyped, packed W ----------------

__global__ __launch_bounds__(128)
void k_gemm256(const _Float16* __restrict__ A, const _Float16* __restrict__ Wp,
               const float* __restrict__ bias, int M, float* __restrict__ out) {
  int wid  = (blockIdx.x * 128 + threadIdx.x) >> 5;
  int lane = threadIdx.x & 31;
  int total = (M >> 4) * 16;
  if (wid >= total) return;
  int m0 = (wid >> 4) * 16, tn = wid & 15, n0 = tn * 16;
  int hs = lane >> 4, mr = lane & 15;
  v8f acc = {};
  const _Float16* Ab = A + (size_t)(m0 + mr) * 256 + hs * 16;
  const _Float16* Wb = Wp + (((size_t)tn * 8) * 32 + lane) * 16;
#pragma unroll
  for (int kb = 0; kb < 8; ++kb) {
    v16h af = load_a_frag(Ab + kb * 32);
    v16h bf = *(const v16h*)(Wb + (size_t)kb * 32 * 16);
    acc = __builtin_amdgcn_wmma_f32_16x16x32_f16(false, af, false, bf, (short)0, acc, false, false);
  }
#pragma unroll
  for (int i = 0; i < 8; ++i) {
    int row = m0 + i + hs * 8, col = n0 + mr;
    float v = acc[i];
    if (bias) v += bias[col];
    out[(size_t)row * 256 + col] = v;
  }
}

// ---------------- typed WMMA GEMM: accumulate all T=4 weights, select per row ----
// mode 0: plain   mode 1: tanh   mode 2: skip-blend with xold (in-place safe)

__global__ __launch_bounds__(128)
void k_typed_gemm256(const _Float16* __restrict__ A, const _Float16* __restrict__ Wp,
                     const float* __restrict__ bias, const int* __restrict__ types,
                     int M, float* out, int mode, const float* xold,
                     const float* __restrict__ skipP) {
  int wid  = (blockIdx.x * 128 + threadIdx.x) >> 5;
  int lane = threadIdx.x & 31;
  int total = (M >> 4) * 16;
  if (wid >= total) return;
  int m0 = (wid >> 4) * 16, tn = wid & 15, n0 = tn * 16;
  int hs = lane >> 4, mr = lane & 15;
  v8f a0 = {}, a1 = {}, a2 = {}, a3 = {};
  const _Float16* Ab = A + (size_t)(m0 + mr) * 256 + hs * 16;
  const _Float16* Wb = Wp + (((size_t)tn * 8) * 32 + lane) * 16;
#pragma unroll
  for (int kb = 0; kb < 8; ++kb) {
    v16h af = load_a_frag(Ab + kb * 32);
    const _Float16* wk = Wb + (size_t)kb * 32 * 16;
    v16h b0 = *(const v16h*)(wk);
    v16h b1 = *(const v16h*)(wk + (size_t)1 * 65536);
    v16h b2 = *(const v16h*)(wk + (size_t)2 * 65536);
    v16h b3 = *(const v16h*)(wk + (size_t)3 * 65536);
    a0 = __builtin_amdgcn_wmma_f32_16x16x32_f16(false, af, false, b0, (short)0, a0, false, false);
    a1 = __builtin_amdgcn_wmma_f32_16x16x32_f16(false, af, false, b1, (short)0, a1, false, false);
    a2 = __builtin_amdgcn_wmma_f32_16x16x32_f16(false, af, false, b2, (short)0, a2, false, false);
    a3 = __builtin_amdgcn_wmma_f32_16x16x32_f16(false, af, false, b3, (short)0, a3, false, false);
  }
#pragma unroll
  for (int i = 0; i < 8; ++i) {
    int row = m0 + i + hs * 8, col = n0 + mr;
    int ty = types[row];
    float v = (ty == 0 ? a0[i] : ty == 1 ? a1[i] : ty == 2 ? a2[i] : a3[i]);
    v += bias[ty * 256 + col];
    if (mode == 1) v = tanhf(v);
    size_t idx = (size_t)row * 256 + col;
    if (mode == 2) {
      float sk = 1.0f / (1.0f + expf(-skipP[ty]));
      v = v * sk + xold[idx] * (1.0f - sk);
    }
    out[idx] = v;
  }
}

// ---------------- edge pass A: attention scores + segment max --------------

__global__ __launch_bounds__(32)
void k_edge_att(const int* __restrict__ src, const int* __restrict__ tgt,
                const int* __restrict__ etype, const int* __restrict__ etime,
                const int* __restrict__ ntype,
                const float* __restrict__ Kn, const float* __restrict__ Qn,
                const float* __restrict__ rteK,
                const float* __restrict__ relAtt, const float* __restrict__ relPri,
                float* __restrict__ attE, unsigned* __restrict__ amaxKey) {
  __shared__ float kld[256];
  __shared__ float qld[256];
  int e = blockIdx.x, lane = threadIdx.x;
  int s = src[e], t = tgt[e], r = etype[e], tm = etime[e];
  int st = ntype[s];
  const float* rk = rteK + ((size_t)st * HGT_ML + tm) * 256;
  const float* kn = Kn + (size_t)s * 256;
  const float* qn = Qn + (size_t)t * 256;
#pragma unroll
  for (int i = 0; i < 8; ++i) {
    int c = lane + i * 32;
    kld[c] = kn[c] + rk[c];
    qld[c] = qn[c];
  }
  __syncthreads();
  const float invs = 0.17677669529663687f; // 1/sqrt(32)
  for (int h = 0; h < 8; ++h) {
    const float* Am = relAtt + ((size_t)r * 8 + h) * 1024;
    float kr = 0.f;
#pragma unroll
    for (int i = 0; i < 32; ++i) kr += kld[h * 32 + i] * Am[i * 32 + lane];
    float p = qld[h * 32 + lane] * kr;
#pragma unroll
    for (int off = 16; off > 0; off >>= 1) p += __shfl_xor(p, off, 32);
    if (lane == 0) {
      float a = p * relPri[r * 8 + h] * invs;
      attE[(size_t)e * 8 + h] = a;
      unsigned bits = __float_as_uint(a);
      unsigned key = (bits & 0x80000000u) ? ~bits : (bits | 0x80000000u);
      atomicMax(&amaxKey[t * 8 + h], key);
    }
  }
}

// ---------------- edge pass B: exp + segment sum --------------

__global__ void k_edge_exp(const int* __restrict__ tgt, float* __restrict__ attE,
                           const unsigned* __restrict__ amaxKey, float* __restrict__ denom) {
  int idx = blockIdx.x * blockDim.x + threadIdx.x;
  if (idx >= HGT_E * 8) return;
  int e = idx >> 3, h = idx & 7;
  int t = tgt[e];
  unsigned key = amaxKey[t * 8 + h];
  unsigned bits = (key & 0x80000000u) ? (key ^ 0x80000000u) : ~key;
  float amax = __uint_as_float(bits);
  float ex = expf(attE[idx] - amax);
  attE[idx] = ex;
  atomicAdd(&denom[t * 8 + h], ex);
}

// ---------------- edge pass C: alpha, message matmul, aggregate --------------

__global__ __launch_bounds__(32)
void k_edge_agg(const int* __restrict__ src, const int* __restrict__ tgt,
                const int* __restrict__ etype, const int* __restrict__ etime,
                const int* __restrict__ ntype,
                const float* __restrict__ Vn, const float* __restrict__ rteV,
                const float* __restrict__ relMsg, const float* __restrict__ attE,
                const float* __restrict__ denom,
                float* __restrict__ alphaOut, float* __restrict__ agg) {
  __shared__ float vld[256];
  int e = blockIdx.x, lane = threadIdx.x;
  int s = src[e], t = tgt[e], r = etype[e], tm = etime[e];
  int st = ntype[s];
  const float* rv = rteV + ((size_t)st * HGT_ML + tm) * 256;
  const float* vn = Vn + (size_t)s * 256;
#pragma unroll
  for (int i = 0; i < 8; ++i) { int c = lane + i * 32; vld[c] = vn[c] + rv[c]; }
  __syncthreads();
  for (int h = 0; h < 8; ++h) {
    float alpha = attE[(size_t)e * 8 + h] / denom[t * 8 + h];
    if (lane == 0) alphaOut[(size_t)e * 8 + h] = alpha;
    const float* Mm = relMsg + ((size_t)r * 8 + h) * 1024;
    float vr = 0.f;
#pragma unroll
    for (int i = 0; i < 32; ++i) vr += vld[h * 32 + i] * Mm[i * 32 + lane];
    atomicAdd(&agg[(size_t)t * 256 + h * 32 + lane], alpha * vr);
  }
}

// ---------------- launcher ----------------

extern "C" void kernel_launch(void* const* d_in, const int* in_sizes, int n_in,
                              void* d_out, int out_size, void* d_ws, size_t ws_size,
                              hipStream_t stream) {
  (void)in_sizes; (void)n_in; (void)out_size; (void)ws_size;
  const float* node_feature = (const float*)d_in[0];
  const int*   node_type    = (const int*)d_in[1];
  const int*   edge_time    = (const int*)d_in[2];
  const int*   edge_index   = (const int*)d_in[3];
  const int*   edge_type    = (const int*)d_in[4];
  const float* adapt_W = (const float*)d_in[5];
  const float* adapt_b = (const float*)d_in[6];
  const float* k_W = (const float*)d_in[7];
  const float* k_b = (const float*)d_in[8];
  const float* q_W = (const float*)d_in[9];
  const float* q_b = (const float*)d_in[10];
  const float* v_W = (const float*)d_in[11];
  const float* v_b = (const float*)d_in[12];
  const float* a_W = (const float*)d_in[13];
  const float* a_b = (const float*)d_in[14];
  const float* rel_pri = (const float*)d_in[15];
  const float* rel_att = (const float*)d_in[16];
  const float* rel_msg = (const float*)d_in[17];
  const float* skip    = (const float*)d_in[18];
  const float* rte_W   = (const float*)d_in[19];
  const float* rte_b   = (const float*)d_in[20];
  const float* rte_emb = (const float*)d_in[21];
  const int* src = edge_index;
  const int* tgt = edge_index + HGT_E;

  char* w = (char*)d_ws;
  size_t off = 0;
  auto alloc = [&](size_t bytes) -> char* {
    char* p = w + off;
    off += (bytes + 255) & ~(size_t)255;
    return p;
  };
  _Float16* a16    = (_Float16*)alloc((size_t)HGT_N * 256 * 2);
  _Float16* adW16  = (_Float16*)alloc((size_t)HGT_T * 65536 * 2);
  _Float16* kW16   = (_Float16*)alloc((size_t)2 * HGT_T * 65536 * 2);
  _Float16* qW16   = (_Float16*)alloc((size_t)2 * HGT_T * 65536 * 2);
  _Float16* vW16   = (_Float16*)alloc((size_t)2 * HGT_T * 65536 * 2);
  _Float16* aW16   = (_Float16*)alloc((size_t)2 * HGT_T * 65536 * 2);
  _Float16* rW16   = (_Float16*)alloc((size_t)2 * 65536 * 2);
  _Float16* emb16  = (_Float16*)alloc((size_t)HGT_ML * 256 * 2);
  _Float16* proj16 = (_Float16*)alloc((size_t)HGT_ML * 256 * 2);
  _Float16* agg16  = (_Float16*)alloc((size_t)HGT_N * 256 * 2);
  float* proj   = (float*)alloc((size_t)HGT_ML * 256 * 4);
  float* rteK   = (float*)alloc((size_t)HGT_T * HGT_ML * 256 * 4);
  float* rteV   = (float*)alloc((size_t)HGT_T * HGT_ML * 256 * 4);
  float* Xa     = (float*)alloc((size_t)HGT_N * 256 * 4);
  float* Qn     = (float*)alloc((size_t)HGT_N * 256 * 4);
  float* Kn     = (float*)alloc((size_t)HGT_N * 256 * 4);
  float* Vn     = (float*)alloc((size_t)HGT_N * 256 * 4);
  float* attE   = (float*)alloc((size_t)HGT_E * 8 * 4);
  unsigned* amaxKey = (unsigned*)alloc((size_t)HGT_N * 8 * 4);
  float* denom  = (float*)alloc((size_t)HGT_N * 8 * 4);
  float* agg    = (float*)alloc((size_t)HGT_N * 256 * 4);

  auto cvt = [&](const float* s, _Float16* d, int n) {
    int blocks = (n + 255) / 256; if (blocks > 4096) blocks = 4096;
    k_cvt_f16<<<blocks, 256, 0, stream>>>(s, d, n);
  };
  auto packB = [&](const float* s, _Float16* d, int nmats) {
    int n = nmats * 65536;
    int blocks = (n + 255) / 256; if (blocks > 4096) blocks = 4096;
    k_pack_b<<<blocks, 256, 0, stream>>>(s, d, nmats);
  };
  auto zero = [&](unsigned* p, int n) {
    int blocks = (n + 255) / 256; if (blocks > 4096) blocks = 4096;
    k_zero32<<<blocks, 256, 0, stream>>>(p, n);
  };

  // one-time (per call) weight pack: f32 -> f16 B-fragment layout
  packB(adapt_W, adW16, HGT_T);
  packB(k_W, kW16, 2 * HGT_T);
  packB(q_W, qW16, 2 * HGT_T);
  packB(v_W, vW16, 2 * HGT_T);
  packB(a_W, aW16, 2 * HGT_T);
  packB(rte_W, rW16, 2);
  cvt(rte_emb, emb16, HGT_ML * 256);
  cvt(node_feature, a16, HGT_N * 256);

  const int blkN   = (HGT_N / 16 * 16 + 3) / 4;   // waves = (M/16)*16 tiles, 4 waves/block
  const int blk240 = (HGT_ML / 16 * 16 + 3) / 4;

  // adapt: x = tanh(typed_linear(feat))
  k_typed_gemm256<<<blkN, 128, 0, stream>>>(a16, adW16, adapt_b, node_type,
                                            HGT_N, Xa, 1, nullptr, nullptr);

  float* alphaOut = (float*)d_out;
  float* xout = alphaOut + (size_t)HGT_E * 8;

  for (int l = 0; l < 2; ++l) {
    cvt(Xa, a16, HGT_N * 256);

    // rte_proj = rte_emb @ rte_W[l] + rte_b[l]  (240x256)
    k_gemm256<<<blk240, 128, 0, stream>>>(emb16, rW16 + (size_t)l * 65536,
                                          rte_b + l * 256, HGT_ML, proj);
    cvt(proj, proj16, HGT_ML * 256);
    // rteK[t] = rte_proj @ k_W[l,t]; rteV[t] = rte_proj @ v_W[l,t]
    for (int t = 0; t < HGT_T; ++t) {
      k_gemm256<<<blk240, 128, 0, stream>>>(proj16, kW16 + ((size_t)l * HGT_T + t) * 65536,
                                            nullptr, HGT_ML, rteK + (size_t)t * HGT_ML * 256);
      k_gemm256<<<blk240, 128, 0, stream>>>(proj16, vW16 + ((size_t)l * HGT_T + t) * 65536,
                                            nullptr, HGT_ML, rteV + (size_t)t * HGT_ML * 256);
    }

    // per-node Q/K/V typed linears (WMMA)
    k_typed_gemm256<<<blkN, 128, 0, stream>>>(a16, qW16 + (size_t)l * HGT_T * 65536,
                                              q_b + l * HGT_T * 256, node_type,
                                              HGT_N, Qn, 0, nullptr, nullptr);
    k_typed_gemm256<<<blkN, 128, 0, stream>>>(a16, kW16 + (size_t)l * HGT_T * 65536,
                                              k_b + l * HGT_T * 256, node_type,
                                              HGT_N, Kn, 0, nullptr, nullptr);
    k_typed_gemm256<<<blkN, 128, 0, stream>>>(a16, vW16 + (size_t)l * HGT_T * 65536,
                                              v_b + l * HGT_T * 256, node_type,
                                              HGT_N, Vn, 0, nullptr, nullptr);

    zero(amaxKey, HGT_N * 8);
    zero((unsigned*)denom, HGT_N * 8);
    zero((unsigned*)agg, HGT_N * 256);

    k_edge_att<<<HGT_E, 32, 0, stream>>>(src, tgt, edge_type, edge_time, node_type,
                                         Kn, Qn, rteK,
                                         rel_att + (size_t)l * 65536,
                                         rel_pri + (size_t)l * 64, attE, amaxKey);
    k_edge_exp<<<(HGT_E * 8 + 255) / 256, 256, 0, stream>>>(tgt, attE, amaxKey, denom);
    k_edge_agg<<<HGT_E, 32, 0, stream>>>(src, tgt, edge_type, edge_time, node_type,
                                         Vn, rteV, rel_msg + (size_t)l * 65536,
                                         attE, denom, alphaOut, agg);

    // gelu(agg) -> f16, then typed a_W + skip blend
    {
      int n = HGT_N * 256;
      int blocks = (n + 255) / 256; if (blocks > 4096) blocks = 4096;
      k_gelu_cvt<<<blocks, 256, 0, stream>>>(agg, agg16, n);
    }
    float* xdst = (l == 0) ? Xa : xout;
    k_typed_gemm256<<<blkN, 128, 0, stream>>>(agg16, aW16 + (size_t)l * HGT_T * 65536,
                                              a_b + l * HGT_T * 256, node_type,
                                              HGT_N, xdst, 2, Xa, skip + l * HGT_T);
  }
}
